// PBGLayer_10634339024955
// MI455X (gfx1250) — matrix operand order repaired and checked by reading, not compiled, not run on hardware
//
#include <hip/hip_runtime.h>
#include <cmath>

// ---------------------------------------------------------------------------
// PBG layer:  out[b,m] = mean_t erf( (sum_n s[b,n,t] * W[m,n] + bias[m]) / sqrt(2N) )
// with s ~ 2*Bernoulli(0.5+0.5*x[b,n]) - 1.
//
// CDNA5 WMMA core: s is exactly representable in f16 (+-1.0h, pure bit
// packing), weights packed fp32->fp16 once into d_ws (L2-resident, 0.5 MB),
// f32 accumulation via V_WMMA_F32_16X16X32_F16. Sign tile generated once per
// block into LDS; RNG draws 4 Bernoulli samples per 32-bit hash (byte vs
// 8-bit threshold) since p depends only on (b,n), not t.
// ---------------------------------------------------------------------------

typedef __attribute__((ext_vector_type(16))) _Float16    v16h;
typedef __attribute__((ext_vector_type(8)))  float       v8f;
typedef __attribute__((ext_vector_type(4)))  unsigned int v4u;

#define B_DIM 512
#define N_DIM 512
#define M_DIM 512
#define T_DIM 100
#define T_TILES 7          // ceil(100/16)

union FragU { v16h h; v4u q[2]; };

__device__ __forceinline__ uint32_t hash32(uint32_t x) {
  // lowbias32
  x ^= x >> 16; x *= 0x7feb352dU;
  x ^= x >> 15; x *= 0x846ca68bU;
  x ^= x >> 16;
  return x;
}

__global__ void zero_out_kernel(float* __restrict__ out, int n) {
  int i = blockIdx.x * blockDim.x + threadIdx.x;
  if (i < n) out[i] = 0.0f;
}

__global__ void pack_weight_kernel(const float* __restrict__ w,
                                   _Float16* __restrict__ wh, int n) {
  int i = blockIdx.x * blockDim.x + threadIdx.x;
  if (i < n) wh[i] = (_Float16)w[i];
}

__global__ __launch_bounds__(256)
void pbg_wmma_kernel(const float* __restrict__ input,
                     const _Float16* __restrict__ wh,
                     const float* __restrict__ bias,
                     float* __restrict__ out) {
  __shared__ unsigned char s_th8[N_DIM];     // 8-bit Bernoulli thresholds
  __shared__ _Float16 s_s[16 * N_DIM];       // sign tile: 16 t-rows x 512 n

  const int tid = threadIdx.x;
  const int b   = blockIdx.y;
  const int t0  = blockIdx.x * 16;

  // --- stage 8-bit thresholds for input row b -----------------------------
  for (int n = tid; n < N_DIM; n += 256) {
    float p = 0.5f + 0.5f * input[b * N_DIM + n];
    int t8 = (int)(p * 256.0f + 0.5f);
    s_th8[n] = (unsigned char)(t8 > 255 ? 255 : t8);
  }
  __syncthreads();

  // --- generate 16x512 sign tile: 4 samples (one per t in quad) per hash --
  {
    const int tq = tid & 3;                  // t-quad: covers t = tq*4 .. +3
    const int nb = (tid >> 2) * 8;           // 64 groups x 8 n values
    const uint32_t tqg = (uint32_t)(blockIdx.x * 4 + tq);  // global t-quad id
#pragma unroll
    for (int i = 0; i < 4; ++i) {
      const int n0 = nb + 2 * i, n1 = n0 + 1;
      const uint32_t u0 = hash32((((uint32_t)((b << 9) | n0)) << 5) | tqg);
      const uint32_t u1 = hash32((((uint32_t)((b << 9) | n1)) << 5) | tqg);
      const uint32_t th0 = s_th8[n0], th1 = s_th8[n1];
#pragma unroll
      for (int j = 0; j < 4; ++j) {
        const uint32_t h0 = (((u0 >> (8 * j)) & 255u) < th0) ? 0x3C00u : 0xBC00u;
        const uint32_t h1 = (((u1 >> (8 * j)) & 255u) < th1) ? 0x3C00u : 0xBC00u;
        const int t_local = tq * 4 + j;
        *(uint32_t*)&s_s[t_local * N_DIM + n0] = h0 | (h1 << 16);
      }
    }
  }
  __syncthreads();

  const int wave     = tid >> 5;
  const int lane     = tid & 31;
  const int laneM    = lane & 15;
  const int laneHalf = lane >> 4;
  const float invs   = 0.03125f;             // 1/sqrt(2*512) = 1/32

#pragma unroll
  for (int j = 0; j < 4; ++j) {              // 8 waves x 4 tiles = all 512 m
    const int mt    = wave * 4 + j;
    const int m_col = mt * 16 + laneM;
    const _Float16* wrow = wh + m_col * N_DIM;
    const float bmy = bias[m_col];
    v8f acc = {};

    for (int k = 0; k < N_DIM; k += 32) {
      // A fragment (16x32 f16, documented striping: lanes 0-15 K 0-7/16-23,
      // lanes 16-31 K 8-15/24-31), rows = t, from the LDS sign tile
      FragU fa;
      const _Float16* abase = &s_s[laneM * N_DIM + k + laneHalf * 8];
      fa.q[0] = *(const v4u*)(abase);
      fa.q[1] = *(const v4u*)(abase + 16);
      // B fragment (32x16 f16): lane = column m, lanes 0-15 K 0-7/16-23,
      // lanes 16-31 K 8-15/24-31; packed halves from L2
      FragU fb;
      const _Float16* bbase = wrow + k + laneHalf * 8;
      fb.q[0] = *(const v4u*)(bbase);
      fb.q[1] = *(const v4u*)(bbase + 16);

      acc = __builtin_amdgcn_wmma_f32_16x16x32_f16(
          /*neg_a=*/false, fa.h, /*neg_b=*/false, fb.h,
          /*c_mod=*/(short)0, acc, /*reuse_a=*/false, /*reuse_b=*/false);
    }

    // epilogue: bias, scale, erf, mask t<T, reduce over the 16 t-rows
    float sum = 0.0f;
#pragma unroll
    for (int r = 0; r < 8; ++r) {
      const int tg = t0 + r + laneHalf * 8;  // C/D layout: VGPR r -> M=r (+8)
      float z = (acc[r] + bmy) * invs;
      float e = erff(z);
      sum += (tg < T_DIM) ? e : 0.0f;
    }
    sum += __shfl_xor(sum, 16, 32);          // combine the two half-waves
    if (laneHalf == 0) {
      atomicAdd(&out[b * M_DIM + m_col], sum * (1.0f / T_DIM));
    }
  }
}

extern "C" void kernel_launch(void* const* d_in, const int* in_sizes, int n_in,
                              void* d_out, int out_size, void* d_ws, size_t ws_size,
                              hipStream_t stream) {
  const float* input  = (const float*)d_in[0];   // [B, N]
  const float* weight = (const float*)d_in[1];   // [M, N]
  const float* bias   = (const float*)d_in[2];   // [M]
  float* out = (float*)d_out;                    // [B, M]
  _Float16* wh = (_Float16*)d_ws;                // packed fp16 weights (512 KB)

  const int outElems = B_DIM * M_DIM;
  const int wElems   = M_DIM * N_DIM;

  zero_out_kernel<<<(outElems + 255) / 256, 256, 0, stream>>>(out, outElems);
  pack_weight_kernel<<<(wElems + 255) / 256, 256, 0, stream>>>(weight, wh, wElems);
  pbg_wmma_kernel<<<dim3(T_TILES, B_DIM), 256, 0, stream>>>(input, wh, bias, out);
}
